// CrossAttention_58660663328896
// MI455X (gfx1250) — compile-verified, matrix operand order checked by reading, not run on hardware
//
#include <hip/hip_runtime.h>
#include <hip/hip_bf16.h>

#define USE_ASYNC_LDS 1

typedef unsigned short u16;
typedef __attribute__((ext_vector_type(16))) __bf16 v16bf;
typedef __attribute__((ext_vector_type(8)))  float  v8f;
typedef __attribute__((ext_vector_type(8)))  unsigned short su8;

union Frag16 { v16bf v; su8 h[2]; };

__device__ __forceinline__ u16 f32_to_bf16u(float f) {
  unsigned u = __builtin_bit_cast(unsigned, f);
  unsigned r = u + 0x7FFFu + ((u >> 16) & 1u);   // round-to-nearest-even
  return (u16)(r >> 16);
}

// Async memory->LDS copy of 16 bytes (CDNA5 GLOBAL_LOAD_ASYNC_TO_LDS_B128,
// tracked by ASYNCcnt). LDS address = flat address low 32 bits.
__device__ __forceinline__ void async_copy_b128(const u16* gsrc, u16* lds_dst) {
#if USE_ASYNC_LDS
  unsigned lds_off = (unsigned)(unsigned long long)lds_dst;
  asm volatile("global_load_async_to_lds_b128 %0, %1, off"
               :: "v"(lds_off), "v"(gsrc) : "memory");
#else
  *(su8*)lds_dst = *(const su8*)gsrc;
#endif
}

__device__ __forceinline__ void wait_async_all() {
#if USE_ASYNC_LDS
  asm volatile("s_wait_asynccnt 0x0" ::: "memory");
#endif
}

// ---------------------------------------------------------------------------
// f32 -> bf16 conversion (grid-stride)
// ---------------------------------------------------------------------------
__global__ __launch_bounds__(256) void cvt_f32_bf16(const float* __restrict__ in,
                                                    u16* __restrict__ out, long n) {
  long i = (long)blockIdx.x * blockDim.x + threadIdx.x;
  long stride = (long)gridDim.x * blockDim.x;
  for (; i < n; i += stride) out[i] = f32_to_bf16u(in[i]);
}

// ---------------------------------------------------------------------------
// bf16 WMMA GEMM:  C[M,N] = scale * (A[M,K] @ B[K,N]) (+ bias)
// Tile: 256(M) x 64(N), K-step 32. 256 threads = 8 waves; wave w owns a
// 32x64 sub-tile: B-frags loaded once, 2 A-frags, 8 WMMAs per k-step.
// ---------------------------------------------------------------------------
template<bool BF16_OUT>
__global__ __launch_bounds__(256)
void gemm_bf16_wmma(const u16* __restrict__ A, const u16* __restrict__ B,
                    void* __restrict__ Cout, const float* __restrict__ bias,
                    int M, int N, int K, float scale)
{
  __shared__ u16 As[256][40];   // row-major, padded (80B rows, 16B aligned)
  __shared__ u16 BsT[64][40];   // B transposed: [n][k], padded

  const int tid  = threadIdx.x;
  const int wave = tid >> 5;
  const int lane = tid & 31;
  const int l15  = lane & 15;
  const int sk   = (lane & 16) ? 8 : 0;   // K sub-offset per ISA 16-bit layout
  const int m0   = blockIdx.y * 256;
  const int n0   = blockIdx.x * 64;

  v8f acc[2][4] = {};

  for (int k0 = 0; k0 < K; k0 += 32) {
    __syncthreads();
    { // stage A tile 256x32: one row (32 u16 = 4x b128) per thread, async
      const u16* g = A + (long)(m0 + tid) * K + k0;
#pragma unroll
      for (int i = 0; i < 4; ++i) async_copy_b128(g + 8 * i, &As[tid][8 * i]);
    }
    { // stage B tile 32x64 transposed into BsT[n][k] (register path)
      int kk = tid >> 3, noff = (tid & 7) * 8;
      const u16* g = B + (long)(k0 + kk) * N + n0 + noff;
      __builtin_prefetch(g + (long)32 * N, 0, 0);
#pragma unroll
      for (int i = 0; i < 8; ++i) BsT[noff + i][kk] = g[i];
    }
    wait_async_all();
    __syncthreads();

    Frag16 bf[4];
#pragma unroll
    for (int nb = 0; nb < 4; ++nb) {
      const int col = nb * 16 + l15;
      bf[nb].h[0] = *(const su8*)&BsT[col][sk];
      bf[nb].h[1] = *(const su8*)&BsT[col][16 + sk];
    }
#pragma unroll
    for (int mi = 0; mi < 2; ++mi) {
      Frag16 a;
      const int row = wave * 32 + mi * 16 + l15;
      a.h[0] = *(const su8*)&As[row][sk];
      a.h[1] = *(const su8*)&As[row][16 + sk];
#pragma unroll
      for (int nb = 0; nb < 4; ++nb)
        acc[mi][nb] = __builtin_amdgcn_wmma_f32_16x16x32_bf16(
            false, a.v, false, bf[nb].v, (short)0, acc[mi][nb], false, false);
    }
  }

  // Epilogue: C layout — VGPR r: lanes 0-15 row r, lanes 16-31 row r+8
#pragma unroll
  for (int mi = 0; mi < 2; ++mi)
#pragma unroll
    for (int nb = 0; nb < 4; ++nb) {
      const int n = n0 + nb * 16 + l15;
      const float bv = bias ? bias[n] : 0.0f;
#pragma unroll
      for (int r = 0; r < 8; ++r) {
        const int m = m0 + wave * 32 + mi * 16 + ((lane & 16) ? r + 8 : r);
        const float v = acc[mi][nb][r] * scale + bv;
        if (BF16_OUT) ((u16*)Cout)[(long)m * N + n] = f32_to_bf16u(v);
        else          ((float*)Cout)[(long)m * N + n] = v;
      }
    }
}

// ---------------------------------------------------------------------------
// Flash attention (bf16 in, bf16 out): per block = (b, h, 128 q-rows).
// Wave w owns q-rows w*16..w*16+15. KV swept in chunks of 64.
// Q/K/V layouts: [B*S, H*64] row-major (h-major flatten, matches reference).
// ---------------------------------------------------------------------------
__global__ __launch_bounds__(256)
void flash_attn_wmma(const u16* __restrict__ Qg, const u16* __restrict__ Kg,
                     const u16* __restrict__ Vg, u16* __restrict__ Og,
                     int Sq, int Skv)
{
  __shared__ u16 Qs[128][72];       // q rows x 64 d
  __shared__ u16 Ks[64][72];        // kv rows x 64 d  (B-frag: lane = kv col)
  __shared__ u16 Vt[64][72];        // d rows x 64 kv  (B-frag: lane = d col)
  __shared__ u16 Ps[8][16][72];     // per-wave P tile for C->A layout swap

  const int tid  = threadIdx.x;
  const int wave = tid >> 5;
  const int lane = tid & 31;
  const int l15  = lane & 15;
  const int sk   = (lane & 16) ? 8 : 0;
  const int q0   = blockIdx.x * 128;
  const int h    = blockIdx.y;
  const int b    = blockIdx.z;

  const long baseQ  = ((long)b * Sq + q0) * 1024 + h * 64;
  const long baseKV = ((long)b * Skv) * 1024 + h * 64;

  { // stage Q tile (128x64) via async copies
    int r = tid >> 1, doff = (tid & 1) * 32;
    const u16* g = Qg + baseQ + (long)r * 1024 + doff;
#pragma unroll
    for (int i = 0; i < 4; ++i) async_copy_b128(g + 8 * i, &Qs[r][doff + 8 * i]);
  }

  v8f oacc[4] = {};
  float mrow[8], lrow[8];
#pragma unroll
  for (int r = 0; r < 8; ++r) { mrow[r] = -INFINITY; lrow[r] = 0.0f; }

  for (int kv0 = 0; kv0 < Skv; kv0 += 64) {
    __syncthreads();
    { // stage K chunk rows as-is via async copies
      int r = tid >> 2, doff = (tid & 3) * 16;
      const u16* g = Kg + baseKV + (long)(kv0 + r) * 1024 + doff;
      async_copy_b128(g,     &Ks[r][doff]);
      async_copy_b128(g + 8, &Ks[r][doff + 8]);
    }
    { // stage V chunk transposed: Vt[d][kv] (register path; needs scatter)
      int r = tid >> 2, doff = (tid & 3) * 16;
      const u16* g = Vg + baseKV + (long)(kv0 + r) * 1024 + doff;
#pragma unroll
      for (int i = 0; i < 16; ++i) Vt[doff + i][r] = g[i];
    }
    wait_async_all();   // also covers the pre-loop Q staging on first pass
    __syncthreads();

    // S = Q @ K^T  (16 q-rows x 64 kv-cols per wave)
    v8f sacc[4] = {};
#pragma unroll
    for (int d0 = 0; d0 < 64; d0 += 32) {
      Frag16 a;
      a.h[0] = *(const su8*)&Qs[wave * 16 + l15][d0 + sk];
      a.h[1] = *(const su8*)&Qs[wave * 16 + l15][d0 + 16 + sk];
#pragma unroll
      for (int nb = 0; nb < 4; ++nb) {
        Frag16 bf;
        bf.h[0] = *(const su8*)&Ks[nb * 16 + l15][d0 + sk];
        bf.h[1] = *(const su8*)&Ks[nb * 16 + l15][d0 + 16 + sk];
        sacc[nb] = __builtin_amdgcn_wmma_f32_16x16x32_bf16(
            false, a.v, false, bf.v, (short)0, sacc[nb], false, false);
      }
    }

    // Online softmax: rows live across the 16-lane half-groups (xor<=8 safe)
#pragma unroll
    for (int r = 0; r < 8; ++r) {
      float mx = fmaxf(fmaxf(sacc[0][r], sacc[1][r]),
                       fmaxf(sacc[2][r], sacc[3][r]));
#pragma unroll
      for (int off = 1; off < 16; off <<= 1) mx = fmaxf(mx, __shfl_xor(mx, off, 32));
      const float mnew  = fmaxf(mrow[r], mx);
      const float alpha = __expf(mrow[r] - mnew);
      float psum = 0.0f;
#pragma unroll
      for (int nb = 0; nb < 4; ++nb) {
        float p = __expf(sacc[nb][r] - mnew);
        sacc[nb][r] = p;
        psum += p;
      }
#pragma unroll
      for (int off = 1; off < 16; off <<= 1) psum += __shfl_xor(psum, off, 32);
      lrow[r] = lrow[r] * alpha + psum;
      mrow[r] = mnew;
#pragma unroll
      for (int nb = 0; nb < 4; ++nb) oacc[nb][r] *= alpha;
    }

    // C-layout -> A-layout swap for P through per-wave LDS
#pragma unroll
    for (int r = 0; r < 8; ++r) {
      const int m = (lane & 16) ? r + 8 : r;
#pragma unroll
      for (int nb = 0; nb < 4; ++nb)
        Ps[wave][m][nb * 16 + l15] = f32_to_bf16u(sacc[nb][r]);
    }

    // O += P @ V  (K dim = 64 kv positions)
#pragma unroll
    for (int k0 = 0; k0 < 64; k0 += 32) {
      Frag16 a;
      a.h[0] = *(const su8*)&Ps[wave][l15][k0 + sk];
      a.h[1] = *(const su8*)&Ps[wave][l15][k0 + 16 + sk];
#pragma unroll
      for (int nb = 0; nb < 4; ++nb) {
        Frag16 bf;
        bf.h[0] = *(const su8*)&Vt[nb * 16 + l15][k0 + sk];
        bf.h[1] = *(const su8*)&Vt[nb * 16 + l15][k0 + 16 + sk];
        oacc[nb] = __builtin_amdgcn_wmma_f32_16x16x32_bf16(
            false, a.v, false, bf.v, (short)0, oacc[nb], false, false);
      }
    }
  }

  // Epilogue: O row m, col h*64 + d; divide by softmax denominator
#pragma unroll
  for (int r = 0; r < 8; ++r) {
    const float inv = 1.0f / lrow[r];
    const int m = q0 + wave * 16 + ((lane & 16) ? r + 8 : r);
    const long rowbase = ((long)b * Sq + m) * 1024 + h * 64;
#pragma unroll
    for (int nb = 0; nb < 4; ++nb)
      Og[rowbase + nb * 16 + l15] = f32_to_bf16u(oacc[nb][r] * inv);
  }
}

// ---------------------------------------------------------------------------
// Host-side orchestration
// ---------------------------------------------------------------------------
extern "C" void kernel_launch(void* const* d_in, const int* in_sizes, int n_in,
                              void* d_out, int out_size, void* d_ws, size_t ws_size,
                              hipStream_t stream) {
  const float* x  = (const float*)d_in[0];
  const float* y  = (const float*)d_in[1];
  const float* Wv = (const float*)d_in[2];
  const float* Wk = (const float*)d_in[3];
  const float* Wq = (const float*)d_in[4];
  const float* Wu = (const float*)d_in[5];
  const float* bu = (const float*)d_in[6];
  float* out = (float*)d_out;

  const int B = 4, S = 2048, E = 1024, H = 16, D = 64;
  const long nTok = (long)B * S;                 // 8192
  const float qk_scale = 0.17677669529663689f;   // 1024^-0.25

  u16* ws  = (u16*)d_ws;
  u16* Xb  = ws;                 // [nTok, E]
  u16* Yb  = Xb  + nTok * E;
  u16* Wvb = Yb  + nTok * E;     // [E, E]
  u16* Wkb = Wvb + (long)E * E;
  u16* Wqb = Wkb + (long)E * E;
  u16* Wub = Wqb + (long)E * E;  // [E, D]
  u16* Vb  = Wub + (long)E * D;  // [nTok, E]
  u16* Kb  = Vb  + nTok * E;
  u16* Qb  = Kb  + nTok * E;
  u16* Ob  = Qb  + nTok * E;

  dim3 blk(256);
  cvt_f32_bf16<<<4096, blk, 0, stream>>>(x,  Xb,  nTok * E);
  cvt_f32_bf16<<<4096, blk, 0, stream>>>(y,  Yb,  nTok * E);
  cvt_f32_bf16<<<1024, blk, 0, stream>>>(Wv, Wvb, (long)E * E);
  cvt_f32_bf16<<<1024, blk, 0, stream>>>(Wk, Wkb, (long)E * E);
  cvt_f32_bf16<<<1024, blk, 0, stream>>>(Wq, Wqb, (long)E * E);
  cvt_f32_bf16<<<256,  blk, 0, stream>>>(Wu, Wub, (long)E * D);

  // QKV projections (scale folded into epilogue for Q, K)
  gemm_bf16_wmma<true><<<dim3(E / 64, nTok / 256), blk, 0, stream>>>(
      Xb, Wvb, Vb, nullptr, (int)nTok, E, E, 1.0f);
  gemm_bf16_wmma<true><<<dim3(E / 64, nTok / 256), blk, 0, stream>>>(
      Xb, Wkb, Kb, nullptr, (int)nTok, E, E, qk_scale);
  gemm_bf16_wmma<true><<<dim3(E / 64, nTok / 256), blk, 0, stream>>>(
      Yb, Wqb, Qb, nullptr, (int)nTok, E, E, qk_scale);

  // Flash attention
  flash_attn_wmma<<<dim3(S / 128, H, B), blk, 0, stream>>>(Qb, Kb, Vb, Ob, S, S);

  // Output projection with bias, f32 output
  gemm_bf16_wmma<false><<<dim3(1, nTok / 256), blk, 0, stream>>>(
      Ob, Wub, out, bu, (int)nTok, D, E, 1.0f);
}
